// VNNv3_5832565588041
// MI455X (gfx1250) — matrix operand order repaired and checked by reading, not compiled, not run on hardware
//
#include <hip/hip_runtime.h>
#include <hip/hip_bf16.h>
#include <math.h>

// ---- shapes fixed by the reference setup_inputs() ----
#define B_SZ    64
#define IN_SZ   1024
#define OUT_SZ  1024
#define D_MODEL 64
#define X_KS    32
#define Y_KS    32
#define HIDDEN  256
#define GEN_X   32          // ceil(1024/32)
#define GEN_Y   32
#define GG      (GEN_X*GEN_Y)     // 1024
#define M_ROWS  (B_SZ*GG)         // 65536
#define KTOT    160               // 2*D_MODEL + X_KS, = 5*32 exactly
#define NKT     5                 // K tiles of 32

typedef _Float16 v16h __attribute__((ext_vector_type(16)));
typedef _Float16 h8   __attribute__((ext_vector_type(8)));
typedef float    v8f  __attribute__((ext_vector_type(8)));

// ---------------- prep kernels ----------------

// pos_enc table for p=0..31: pef (f32) and peH (f16), layout [32][64]
__global__ void k_pe(float* __restrict__ pef, _Float16* __restrict__ peH) {
    int tid = blockIdx.x * blockDim.x + threadIdx.x;   // 0..2047
    if (tid >= 32 * 64) return;
    int p = tid >> 6, d = tid & 63;
    int i = d >> 1;
    float invf = __expf(-(2.0f * (float)i / (float)D_MODEL) * __logf(10000.0f));
    float ang = (float)p * invf;
    float v = (d & 1) ? __cosf(ang) : __sinf(ang);
    pef[tid] = v;
    peH[tid] = (_Float16)v;
}

// x (f32) -> xh (f16)
__global__ void k_xh(const float* __restrict__ x, _Float16* __restrict__ xh) {
    int tid = blockIdx.x * blockDim.x + threadIdx.x;
    if (tid < B_SZ * IN_SZ) xh[tid] = (_Float16)x[tid];
}

// per-batch sum of x (for b2 term)
__global__ void k_xsum(const float* __restrict__ x, float* __restrict__ xsum) {
    __shared__ float sm[256];
    int b = blockIdx.x, t = threadIdx.x;
    float s = 0.f;
    for (int i = t; i < IN_SZ; i += 256) s += x[b * IN_SZ + i];
    sm[t] = s; __syncthreads();
    for (int off = 128; off > 0; off >>= 1) {
        if (t < off) sm[t] += sm[t + off];
        __syncthreads();
    }
    if (t == 0) xsum[b] = sm[0];
}

// Repack W1 [160,256] f32 -> per-lane WMMA B fragments, f16.
// Layout: W1p[(((nt*5)+kt)*32 + lane)*16 + e], where
//   n = nt*16 + (lane&15),  k = kt*32 + 8*(lane>>4) + (e&7) + 16*(e>>3)   (B: table K order, matches A)
__global__ void k_w1pack(const float* __restrict__ W1, _Float16* __restrict__ W1p) {
    int idx = blockIdx.x * blockDim.x + threadIdx.x;     // 0..40959
    if (idx >= 16 * NKT * 32 * 16) return;
    int e    = idx & 15;
    int r    = idx >> 4;
    int lane = r & 31;
    int r2   = r >> 5;
    int kt   = r2 % NKT;
    int nt   = r2 / NKT;
    int n = nt * 16 + (lane & 15);
    int k = kt * 32 + 8 * (lane >> 4) + (e & 7) + 16 * (e >> 3);
    W1p[idx] = (_Float16)W1[k * HIDDEN + n];
}

// ---------------- main fused hypernet GEMM (WMMA) ----------------
// H[b,g,h] = tanh(arg(b,g) . W1[:,h] + b1[h]);  sPart[b][rb][h] = sum over the
// 16 rows g of this block of x[b,g]*H[b,g,h].
// arg(b,g) = [pe(c)[64] | pe(a)[64] | x[b, c*32 : c*32+32]], a=g>>5, c=g&31.
__global__ __launch_bounds__(256) void k_hgemm(
        const _Float16* __restrict__ peH, const _Float16* __restrict__ xh,
        const _Float16* __restrict__ W1p, const float* __restrict__ b1,
        const float* __restrict__ x, float* __restrict__ sPart) {
    int bm   = blockIdx.x * 16;          // global row base (b*1024 + grow)
    int b    = bm >> 10;
    int grow = bm & 1023;
    int a    = grow >> 5;                // constant over the 16 rows
    int wave = threadIdx.x >> 5;
    int lane = threadIdx.x & 31;
    int m16  = lane & 15;
    int hi   = lane >> 4;
    int c    = (grow & 31) + m16;        // this lane's A-row window index

    const _Float16* peC = peH + c * 64;
    const _Float16* peA = peH + a * 64;
    const _Float16* xr  = xh + b * IN_SZ + c * 32;

    int nt0 = wave * 2, nt1 = nt0 + 1;
    v8f acc0 = {}; v8f acc1 = {};

#pragma unroll
    for (int kt = 0; kt < NKT; ++kt) {
        // A fragment (16x32 f16): lane m16(+hi) row, K = kt*32 + e%8 + 8*hi + 16*(e/8)
        const _Float16* src = (kt < 2) ? (peC + kt * 32)
                            : (kt < 4) ? (peA + (kt - 2) * 32)
                                       : xr;
        h8 lo = *(const h8*)(src + 8 * hi);
        h8 hq = *(const h8*)(src + 8 * hi + 16);
        v16h A;
#pragma unroll
        for (int e = 0; e < 8; ++e) { A[e] = lo[e]; A[e + 8] = hq[e]; }

        v16h B0 = *(const v16h*)(W1p + ((size_t)(nt0 * NKT + kt) * 32 + lane) * 16);
        v16h B1 = *(const v16h*)(W1p + ((size_t)(nt1 * NKT + kt) * 32 + lane) * 16);

        acc0 = __builtin_amdgcn_wmma_f32_16x16x32_f16(false, A, false, B0,
                                                      (short)0, acc0, false, false);
        acc1 = __builtin_amdgcn_wmma_f32_16x16x32_f16(false, A, false, B1,
                                                      (short)0, acc1, false, false);
    }

    // epilogue: tanh(+b1), weight by x[b, grow+M], reduce 16 rows per column
    int rb = grow >> 4;                  // row-block index within batch (0..63)
    const float* xg = x + b * IN_SZ + grow;
    float xv[8];
#pragma unroll
    for (int r = 0; r < 8; ++r) xv[r] = xg[r + 8 * hi];   // M = r + 8*hi

#pragma unroll
    for (int t = 0; t < 2; ++t) {
        v8f acc = t ? acc1 : acc0;
        int n = wave * 32 + t * 16 + m16;                 // hidden index
        float bias = b1[n];
        float p = 0.f;
#pragma unroll
        for (int r = 0; r < 8; ++r) p += xv[r] * tanhf(acc[r] + bias);
        p += __shfl_xor(p, 16, 32);                       // combine hi halves
        if (hi == 0)
            sPart[((size_t)(b * 64 + rb)) * HIDDEN + n] = p;
    }
}

// s[b][h] = sum_rb sPart[b][rb][h]
__global__ void k_sreduce(const float* __restrict__ sPart, float* __restrict__ s) {
    int tid = blockIdx.x * blockDim.x + threadIdx.x;      // 0..16383
    if (tid >= B_SZ * HIDDEN) return;
    int b = tid >> 8, h = tid & 255;
    float acc = 0.f;
    for (int rb = 0; rb < 64; ++rb)
        acc += sPart[((size_t)(b * 64 + rb)) * HIDDEN + h];
    s[tid] = acc;
}

// out[b][o] = sum_h s[b][h]*W2[h][o] + xsum[b]*b2[o]
__global__ void k_out(const float* __restrict__ s, const float* __restrict__ W2,
                      const float* __restrict__ b2, const float* __restrict__ xsum,
                      float* __restrict__ outBuf) {
    int tid = blockIdx.x * blockDim.x + threadIdx.x;      // 0..65535
    if (tid >= B_SZ * OUT_SZ) return;
    int b = tid >> 10, o = tid & 1023;
    float acc = xsum[b] * b2[o];
    const float* sb = s + b * HIDDEN;
    for (int h = 0; h < HIDDEN; ++h)
        acc += sb[h] * W2[(size_t)h * OUT_SZ + o];
    outBuf[tid] = acc;
}

// bias hypernet + final add: per (b, a): in96=[out[b,a*32:+32] | pe(a)];
// hb=tanh(in96.Wb1+bb1); bias=hb.Wb2+bb2; d_out = out + bias
__global__ __launch_bounds__(256) void k_bias(
        const float* __restrict__ outBuf, const float* __restrict__ pef,
        const float* __restrict__ Wb1, const float* __restrict__ bb1,
        const float* __restrict__ Wb2, const float* __restrict__ bb2,
        float* __restrict__ dout) {
    __shared__ float in96[96];
    __shared__ float hb[HIDDEN];
    int b = blockIdx.x >> 5, a = blockIdx.x & 31, t = threadIdx.x;
    if (t < 96)
        in96[t] = (t < 32) ? outBuf[b * OUT_SZ + a * 32 + t]
                           : pef[a * 64 + (t - 32)];
    __syncthreads();
    float acc = bb1[t];
#pragma unroll 4
    for (int k = 0; k < 96; ++k) acc += in96[k] * Wb1[k * HIDDEN + t];
    hb[t] = tanhf(acc);
    __syncthreads();
    if (t < 32) {
        float bs = bb2[t];
#pragma unroll 4
        for (int h = 0; h < HIDDEN; ++h) bs += hb[h] * Wb2[h * Y_KS + t];
        int o = a * 32 + t;
        dout[b * OUT_SZ + o] = outBuf[b * OUT_SZ + o] + bs;
    }
}

extern "C" void kernel_launch(void* const* d_in, const int* in_sizes, int n_in,
                              void* d_out, int out_size, void* d_ws, size_t ws_size,
                              hipStream_t stream) {
    const float* x   = (const float*)d_in[0];
    // d_in[1] = output_size (assumed 1024 per setup_inputs)
    const float* W1  = (const float*)d_in[2];
    const float* b1  = (const float*)d_in[3];
    const float* W2  = (const float*)d_in[4];
    const float* b2  = (const float*)d_in[5];
    const float* Wb1 = (const float*)d_in[6];
    const float* bb1 = (const float*)d_in[7];
    const float* Wb2 = (const float*)d_in[8];
    const float* bb2 = (const float*)d_in[9];
    float* dout = (float*)d_out;

    char* ws = (char*)d_ws;
    size_t off = 0;
    auto alloc = [&](size_t bytes) { char* p = ws + off; off = (off + bytes + 255) & ~(size_t)255; return p; };
    _Float16* peH   = (_Float16*)alloc(32 * 64 * sizeof(_Float16));
    float*    pef   = (float*)   alloc(32 * 64 * sizeof(float));
    _Float16* xh    = (_Float16*)alloc((size_t)B_SZ * IN_SZ * sizeof(_Float16));
    _Float16* W1p   = (_Float16*)alloc((size_t)16 * NKT * 32 * 16 * sizeof(_Float16));
    float*    xsum  = (float*)   alloc(B_SZ * sizeof(float));
    float*    sPart = (float*)   alloc((size_t)B_SZ * 64 * HIDDEN * sizeof(float));
    float*    sBuf  = (float*)   alloc((size_t)B_SZ * HIDDEN * sizeof(float));
    float*    outB  = (float*)   alloc((size_t)B_SZ * OUT_SZ * sizeof(float));
    (void)ws_size; (void)in_sizes; (void)n_in; (void)out_size;

    k_pe    <<<8,   256, 0, stream>>>(pef, peH);
    k_xh    <<<256, 256, 0, stream>>>(x, xh);
    k_xsum  <<<64,  256, 0, stream>>>(x, xsum);
    k_w1pack<<<160, 256, 0, stream>>>(W1, W1p);
    k_hgemm <<<M_ROWS / 16, 256, 0, stream>>>(peH, xh, W1p, b1, x, sPart);
    k_sreduce<<<64, 256, 0, stream>>>(sPart, sBuf);
    k_out   <<<256, 256, 0, stream>>>(sBuf, W2, b2, xsum, outB);
    k_bias  <<<B_SZ * GEN_X, 256, 0, stream>>>(outB, pef, Wb1, bb1, Wb2, bb2, dout);
}
